// BertSelfAttention_47571057770844
// MI455X (gfx1250) — matrix operand order repaired and checked by reading, not compiled, not run on hardware
//
#include <hip/hip_runtime.h>
#include <hip/hip_bf16.h>

// ---------------------------------------------------------------------------
// DeBERTa-style self-attention for MI455X (gfx1250, wave32, WMMA)
// B=2, S=2048, D=1024, H=16, DH=64
//
// Roofline: dominant traffic is rel_pos (268MB x 2 batches) at 23.3 TB/s ->
// ~25us floor; everything is built to stay at that floor:
//  - never materialize [B,H,S,S] scores (flash/online softmax)
//  - segment einsum collapsed to 2 dots per row (seg_ids in {0,1})
//  - f16 WMMA (v_wmma_f32_16x16x32_f16) with f32 accumulation
//  - rel_pos streamed with non-temporal hint so it can't thrash the 192MB L2
//    that holds the reused k16/vT16/mask/t2 working set.
// ---------------------------------------------------------------------------

typedef _Float16 f16;
typedef _Float16 v16h __attribute__((ext_vector_type(16)));
typedef _Float16 v8h  __attribute__((ext_vector_type(8)));
typedef float    v8f  __attribute__((ext_vector_type(8)));

#define WMMA_F16(a, b, c) \
  __builtin_amdgcn_wmma_f32_16x16x32_f16(false, (a), false, (b), (short)0, (c), false, false)

constexpr int Bc  = 2;
constexpr int Sc  = 2048;
constexpr int Dc  = 1024;
constexpr int Hc  = 16;
constexpr int DHc = 64;

// ---------------------------------------------------------------------------
// f32 -> f16 elementwise convert
// ---------------------------------------------------------------------------
__global__ void cvt_f32_to_f16(const float* __restrict__ in, f16* __restrict__ out, int n) {
  int i = blockIdx.x * blockDim.x + threadIdx.x;
  if (i < n) out[i] = (f16)in[i];
}

// ---------------------------------------------------------------------------
// QKV projection GEMM: out = hs @ W.T (+bias), result stored f16 per-head.
//   A (h16): row-major [B*S, 1024] f16
//   W16    : row-major [1024(out), 1024(in)] f16  -> rows ARE the WMMA B columns
// One wave computes a 16(M) x 64(N) tile, K swept in steps of 32.
// mode 0: q16 [b,h,s,d] (+bq);  mode 1: k16 [b,h,s,d] * 0.125 (no bias);
// mode 2: vT16 [b,h,d,s] (+bv)  (transposed store for the PV GEMM B-operand)
// ---------------------------------------------------------------------------
__global__ __launch_bounds__(32) void qkv_gemm(
    const f16* __restrict__ A, const f16* __restrict__ W,
    const float* __restrict__ bias, f16* __restrict__ out,
    float scl, int mode)
{
  const int lane = threadIdx.x & 31;
  const int hl   = lane >> 4;   // lane half: 0 or 1
  const int ln   = lane & 15;
  const int m0   = blockIdx.x * 16;
  const int n0   = blockIdx.y * 64;

  v8f acc[4] = {};
  const f16* arow = A + (size_t)(m0 + ln) * Dc;

  for (int k0 = 0; k0 < Dc; k0 += 32) {
    // A operand, 16x32 f16: low lanes hold K[0..7],[16..23]; high lanes +8
    v16h a;
    {
      const f16* p = arow + k0 + hl * 8;
      *(v8h*)&a        = *(const v8h*)p;
      ((v8h*)&a)[1]    = *(const v8h*)(p + 16);
    }
#pragma unroll
    for (int sub = 0; sub < 4; ++sub) {
      // B operand 32x16: lane n holds W[n0+n][k0 + hl*8 .. +7] and [k0 + 16 + hl*8 .. +7]
      const f16* bp = W + (size_t)(n0 + sub * 16 + ln) * Dc + k0 + hl * 8;
      v16h bm;
      *(v8h*)&bm     = *(const v8h*)bp;
      ((v8h*)&bm)[1] = *(const v8h*)(bp + 16);
      acc[sub] = WMMA_F16(a, bm, acc[sub]);
    }
  }

#pragma unroll
  for (int sub = 0; sub < 4; ++sub) {
    const int o  = n0 + sub * 16 + ln;      // output feature (column)
    const float bb = bias ? bias[o] : 0.0f;
    const int h = o >> 6, d = o & 63;
#pragma unroll
    for (int r = 0; r < 8; ++r) {
      const int token = m0 + r + hl * 8;    // C-layout: VGPR r -> row r / r+8
      const int b = token >> 11, s = token & (Sc - 1);
      const float val = acc[sub][r] * scl + bb;
      size_t idx;
      if (mode != 2) idx = (((size_t)b * Hc + h) * Sc + s) * DHc + d;  // [b,h,s,d]
      else           idx = (((size_t)b * Hc + h) * DHc + d) * Sc + s;  // [b,h,d,s]
      out[idx] = (f16)val;
    }
  }
}

// ---------------------------------------------------------------------------
// Segment bias: seg_ids only take values {0,1}, so the whole
// einsum('bnih,bjnh->bnij') collapses to two dot products per (b,h,i):
//   t[s] = sum_d (q[b,h,i,d] + b_q_s[h,d]) * seg_table[s, h*64+d]
// ---------------------------------------------------------------------------
__global__ void seg_bias_kernel(
    const f16* __restrict__ q16, const float* __restrict__ seg_table,
    const float* __restrict__ bqs, float* __restrict__ t2)
{
  int row = blockIdx.x * blockDim.x + threadIdx.x;   // row over [B*H*S)
  if (row >= Bc * Hc * Sc) return;
  int h = (row / Sc) & (Hc - 1);
  const f16*   q  = q16 + (size_t)row * DHc;
  const float* bb = bqs + h * DHc;
  const float* s0 = seg_table + h * DHc;
  const float* s1 = seg_table + Dc + h * DHc;
  float a0 = 0.f, a1 = 0.f;
#pragma unroll 8
  for (int d = 0; d < DHc; ++d) {
    float qq = (float)q[d] + bb[d];
    a0 += qq * s0[d];
    a1 += qq * s1[d];
  }
  t2[(size_t)row * 2 + 0] = a0;
  t2[(size_t)row * 2 + 1] = a1;
}

// ---------------------------------------------------------------------------
// Flash attention: one wave owns a 16-row i-tile of one (b,h); streams j in
// tiles of 32 with online softmax. Never materializes S x S scores.
// rel_pos is the zero-reuse bandwidth floor -> non-temporal loads.
// ---------------------------------------------------------------------------
__global__ __launch_bounds__(128) void attn_kernel(
    const f16* __restrict__ q16, const f16* __restrict__ k16,
    const f16* __restrict__ vT16,
    const float* __restrict__ rel_pos, const float* __restrict__ amask,
    const int* __restrict__ seg_ids, const float* __restrict__ t2,
    float* __restrict__ out)
{
  // per-wave p bounce buffer: 16 rows x 32 cols f16, row stride 40 (80B)
  // -> 16 distinct banks for the A-layout b128 reads, 16B aligned.
  __shared__ __align__(16) f16 pbuf[4][16 * 40];

  const int lane = threadIdx.x & 31;
  const int w    = threadIdx.x >> 5;
  const int hl   = lane >> 4;
  const int ln   = lane & 15;
  const int h    = blockIdx.y;
  const int b    = blockIdx.z;
  const int i0   = (blockIdx.x * 4 + w) * 16;
  const size_t bh = (size_t)b * Hc + h;

  // q A-operands for K-steps d=[0,32) and d=[32,64); loaded once.
  v16h qa0, qa1;
  {
    const f16* qp = q16 + (bh * Sc + i0 + ln) * DHc;
    const f16* p0 = qp + hl * 8;
    *(v8h*)&qa0     = *(const v8h*)p0;
    ((v8h*)&qa0)[1] = *(const v8h*)(p0 + 16);
    const f16* p1 = qp + 32 + hl * 8;
    *(v8h*)&qa1     = *(const v8h*)p1;
    ((v8h*)&qa1)[1] = *(const v8h*)(p1 + 16);
  }

  // per-row segment bias values for this lane's 8 rows (C-layout half-wave)
  float t0v[8], t1v[8];
#pragma unroll
  for (int r = 0; r < 8; ++r) {
    size_t row = bh * Sc + i0 + r + hl * 8;
    t0v[r] = t2[row * 2 + 0];
    t1v[r] = t2[row * 2 + 1];
  }

  float m[8], l[8];
#pragma unroll
  for (int r = 0; r < 8; ++r) { m[r] = -1e30f; l[r] = 0.0f; }
  v8f ctx[4] = {};

  const float* rp_base = rel_pos + (size_t)h * Sc * Sc;  // [h,i,j]  (streamed, NT)
  const float* am_base = amask + (size_t)b * Sc * Sc;    // [b,0,i,j] (reused by 16 heads)
  const int*   sg      = seg_ids + (size_t)b * Sc;

  for (int j0 = 0; j0 < Sc; j0 += 32) {
    // ---- scores: q . k^T for two 16-wide j sub-tiles ----
    v8f s[2] = {};
#pragma unroll
    for (int c = 0; c < 2; ++c) {
      const f16* kp = k16 + (bh * Sc + j0 + c * 16 + ln) * DHc + hl * 8;
      v16h kb0, kb1;
      *(v8h*)&kb0     = *(const v8h*)kp;
      ((v8h*)&kb0)[1] = *(const v8h*)(kp + 16);
      *(v8h*)&kb1     = *(const v8h*)(kp + 32);
      ((v8h*)&kb1)[1] = *(const v8h*)(kp + 48);
      s[c] = WMMA_F16(qa0, kb0, s[c]);
      s[c] = WMMA_F16(qa1, kb1, s[c]);
    }

    // ---- add rel_pos (NT) + mask + segment bias; tile row-max ----
    const int ja = j0 + ln, jb = j0 + 16 + ln;
    const int sga = sg[ja], sgb = sg[jb];
    float p0[8], p1[8], tmax[8];
#pragma unroll
    for (int r = 0; r < 8; ++r) {
      const int i = i0 + r + hl * 8;
      float rp0 = __builtin_nontemporal_load(&rp_base[(size_t)i * Sc + ja]);
      float rp1 = __builtin_nontemporal_load(&rp_base[(size_t)i * Sc + jb]);
      float v0 = s[0][r] + rp0 + am_base[(size_t)i * Sc + ja] + (sga ? t1v[r] : t0v[r]);
      float v1 = s[1][r] + rp1 + am_base[(size_t)i * Sc + jb] + (sgb ? t1v[r] : t0v[r]);
      p0[r] = v0; p1[r] = v1;
      float mx = fmaxf(v0, v1);
#pragma unroll
      for (int msk = 8; msk >= 1; msk >>= 1)   // reduce within 16-lane half
        mx = fmaxf(mx, __shfl_xor(mx, msk, 32));
      tmax[r] = mx;
    }

    // ---- online softmax update ----
#pragma unroll
    for (int r = 0; r < 8; ++r) {
      float mn   = fmaxf(m[r], tmax[r]);
      float corr = __expf(m[r] - mn);
      m[r] = mn;
      float e0 = __expf(p0[r] - mn), e1 = __expf(p1[r] - mn);
      p0[r] = e0; p1[r] = e1;
      float rs = e0 + e1;
#pragma unroll
      for (int msk = 8; msk >= 1; msk >>= 1)
        rs += __shfl_xor(rs, msk, 32);
      l[r] = l[r] * corr + rs;
#pragma unroll
      for (int dt = 0; dt < 4; ++dt) ctx[dt][r] *= corr;
    }

    // ---- C-layout -> A-layout transpose of p via wave-private LDS ----
#pragma unroll
    for (int r = 0; r < 8; ++r) {
      const int il = r + hl * 8;
      pbuf[w][il * 40 + ln]      = (f16)p0[r];
      pbuf[w][il * 40 + 16 + ln] = (f16)p1[r];
    }
    asm volatile("s_wait_dscnt 0" ::: "memory");  // LDS in-order per wave; fence compiler too
    v16h pa;
    {
      const f16* pp = &pbuf[w][ln * 40 + hl * 8];
      *(v8h*)&pa     = *(const v8h*)pp;
      ((v8h*)&pa)[1] = *(const v8h*)(pp + 16);
    }

    // ---- ctx += p . v  (B-operand from transposed V: contiguous per lane) ----
#pragma unroll
    for (int dt = 0; dt < 4; ++dt) {
      const f16* vp = vT16 + (bh * DHc + dt * 16 + ln) * Sc + j0 + hl * 8;
      v16h vb;
      *(v8h*)&vb     = *(const v8h*)vp;
      ((v8h*)&vb)[1] = *(const v8h*)(vp + 16);
      ctx[dt] = WMMA_F16(pa, vb, ctx[dt]);
    }
  }

  // ---- normalize and store [B,S,D] f32, coalesced 64B rows ----
#pragma unroll
  for (int r = 0; r < 8; ++r) {
    const float inv = 1.0f / l[r];
    const int i = i0 + r + hl * 8;
    float* op = out + ((size_t)b * Sc + i) * Dc + h * DHc;
#pragma unroll
    for (int dt = 0; dt < 4; ++dt) op[dt * 16 + ln] = ctx[dt][r] * inv;
  }
}

// ---------------------------------------------------------------------------
// Host-side launcher
// ---------------------------------------------------------------------------
extern "C" void kernel_launch(void* const* d_in, const int* in_sizes, int n_in,
                              void* d_out, int out_size, void* d_ws, size_t ws_size,
                              hipStream_t stream) {
  (void)in_sizes; (void)n_in; (void)out_size; (void)ws_size;

  const float* hs     = (const float*)d_in[0];   // [B,S,D]
  const float* amask  = (const float*)d_in[1];   // [B,1,S,S]
  const float* rel    = (const float*)d_in[2];   // [1,H,S,S]
  const int*   seg    = (const int*)  d_in[3];   // [B,S]
  const float* Wq     = (const float*)d_in[4];   // [D,D]
  const float* bq     = (const float*)d_in[5];   // [D]
  const float* Wk     = (const float*)d_in[6];   // [D,D]
  const float* Wv     = (const float*)d_in[7];   // [D,D]
  const float* bv     = (const float*)d_in[8];   // [D]
  const float* segtab = (const float*)d_in[9];   // [2,D]
  const float* bqs    = (const float*)d_in[10];  // [1,H,1,DH]
  float* out = (float*)d_out;

  char* ws = (char*)d_ws;
  f16*   h16  = (f16*)(ws);                 //  8 MB  hidden f16
  f16*   wq16 = (f16*)(ws + 8388608);       //  2 MB
  f16*   wk16 = (f16*)(ws + 10485760);      //  2 MB
  f16*   wv16 = (f16*)(ws + 12582912);      //  2 MB
  f16*   q16  = (f16*)(ws + 14680064);      //  8 MB  [b,h,s,d]
  f16*   k16  = (f16*)(ws + 23068672);      //  8 MB  [b,h,s,d] * 0.125
  f16*   vT16 = (f16*)(ws + 31457280);      //  8 MB  [b,h,d,s]
  float* t2   = (float*)(ws + 39845888);    //  512 KB  [b,h,s,2]

  const int nH = Bc * Sc * Dc;
  const int nW = Dc * Dc;
  cvt_f32_to_f16<<<(nH + 255) / 256, 256, 0, stream>>>(hs, h16, nH);
  cvt_f32_to_f16<<<(nW + 255) / 256, 256, 0, stream>>>(Wq, wq16, nW);
  cvt_f32_to_f16<<<(nW + 255) / 256, 256, 0, stream>>>(Wk, wk16, nW);
  cvt_f32_to_f16<<<(nW + 255) / 256, 256, 0, stream>>>(Wv, wv16, nW);

  dim3 gg(Bc * Sc / 16, Dc / 64);
  qkv_gemm<<<gg, 32, 0, stream>>>(h16, wq16, bq,      q16,  1.0f,   0);
  qkv_gemm<<<gg, 32, 0, stream>>>(h16, wk16, nullptr, k16,  0.125f, 1);  // fold 1/sqrt(64)
  qkv_gemm<<<gg, 32, 0, stream>>>(h16, wv16, bv,      vT16, 1.0f,   2);

  seg_bias_kernel<<<(Bc * Hc * Sc + 255) / 256, 256, 0, stream>>>(q16, segtab, bqs, t2);

  dim3 ga(Sc / 64, Hc, Bc);  // 4 waves/block, one 16-row i-tile per wave
  attn_kernel<<<ga, 128, 0, stream>>>(q16, k16, vT16, rel, amask, seg, t2, out);
}
